// AnomalyGeneration_57483842289819
// MI455X (gfx1250) — matrix operand order repaired and checked by reading, not compiled, not run on hardware
//
#include <hip/hip_runtime.h>

// MI455X / gfx1250, wave32. WMMA bf16 16x16x32 path with hi/lo split for ~fp32 ranking.

typedef __attribute__((ext_vector_type(16))) __bf16 v16bf;
typedef __attribute__((ext_vector_type(8)))  float  v8f;

#define CDIM   128
#define NCODES 4096
#define BATCH  8

__device__ __forceinline__ unsigned hash_u32(unsigned x) {
  x ^= x >> 16; x *= 0x7feb352dU; x ^= x >> 15; x *= 0x846ca68bU; x ^= x >> 16;
  return x;
}

// monotonic float -> unsigned mapping (ascending)
__device__ __forceinline__ unsigned f2ord(float f) {
  unsigned u = __float_as_uint(f);
  return (u & 0x80000000u) ? ~u : (u | 0x80000000u);
}

// -------------------------------------------------------------------------
// 1) copy q_fine / q_coarse to output (unmasked pixels keep q)
// -------------------------------------------------------------------------
__global__ void copy_q_kernel(const float4* __restrict__ qf,
                              const float4* __restrict__ qc,
                              float4* __restrict__ out) {
  const int NF4 = (BATCH * CDIM * 64 * 64) / 4;   // 1048576
  const int NC4 = (BATCH * CDIM * 32 * 32) / 4;   // 262144
  int i = blockIdx.x * blockDim.x + threadIdx.x;
  if (i < NF4)            out[i] = qf[i];
  else if (i < NF4 + NC4) out[i] = qc[i - NF4];
}

// -------------------------------------------------------------------------
// 2) max-pool binarize M and compact masked pixel ids into lists
// -------------------------------------------------------------------------
__global__ void pool_compact_kernel(const float* __restrict__ M,
                                    unsigned* __restrict__ cntF, unsigned* __restrict__ listF,
                                    unsigned* __restrict__ cntC, unsigned* __restrict__ listC) {
  const int NF = BATCH * 64 * 64;   // 32768
  const int NC = BATCH * 32 * 32;   // 8192
  int tid = blockIdx.x * blockDim.x + threadIdx.x;
  if (tid < NF) {
    int b = tid >> 12, p = tid & 4095;
    int y = p >> 6, x = p & 63;
    const float* Mb = M + (size_t)b * 65536;
    bool any = false;
    for (int yy = 0; yy < 4; ++yy)
      for (int xx = 0; xx < 4; ++xx)
        any |= (Mb[(y * 4 + yy) * 256 + (x * 4 + xx)] > 0.0f);
    if (any) { unsigned s = atomicAdd(cntF, 1u); listF[s] = (unsigned)tid; }
  } else if (tid < NF + NC) {
    int t2 = tid - NF;
    int b = t2 >> 10, p = t2 & 1023;
    int y = p >> 5, x = p & 31;
    const float* Mb = M + (size_t)b * 65536;
    bool any = false;
    for (int yy = 0; yy < 8; ++yy)
      for (int xx = 0; xx < 8; ++xx)
        any |= (Mb[(y * 8 + yy) * 256 + (x * 8 + xx)] > 0.0f);
    if (any) { unsigned s = atomicAdd(cntC, 1u); listC[s] = (unsigned)t2; }
  }
}

// -------------------------------------------------------------------------
// 3) split codebooks into bf16 hi/lo + per-code squared norm
//    one block (128 threads) per code; blocks [0,4096)=fine, [4096,8192)=coarse
// -------------------------------------------------------------------------
__global__ __launch_bounds__(128)
void cb_prep_kernel(const float* __restrict__ cbF, const float* __restrict__ cbC,
                    unsigned short* __restrict__ hiF, unsigned short* __restrict__ loF,
                    float* __restrict__ normF,
                    unsigned short* __restrict__ hiC, unsigned short* __restrict__ loC,
                    float* __restrict__ normC) {
  __shared__ float red[128];
  int blk = blockIdx.x;
  int which = blk >> 12;
  int n = blk & (NCODES - 1);
  const float* src      = which ? cbC  : cbF;
  unsigned short* hiA   = which ? hiC  : hiF;
  unsigned short* loA   = which ? loC  : loF;
  float* nrm            = which ? normC : normF;
  int c = threadIdx.x;
  float v = src[(size_t)n * CDIM + c];
  unsigned u  = __float_as_uint(v);
  unsigned hb = u & 0xFFFF0000u;
  float    lo = v - __uint_as_float(hb);
  hiA[(size_t)n * CDIM + c] = (unsigned short)(hb >> 16);
  loA[(size_t)n * CDIM + c] = (unsigned short)(__float_as_uint(lo) >> 16);
  red[c] = v * v;
  __syncthreads();
  for (int s = 64; s > 0; s >>= 1) { if (c < s) red[c] += red[c + s]; __syncthreads(); }
  if (c == 0) nrm[n] = red[0];
}

// -------------------------------------------------------------------------
// 4) fused distance-GEMM (WMMA bf16, hi/lo split) + exact radix rank-select
//    block = 256 threads (8 waves), 16 masked pixels per block, all 4096 codes
//    LDS: 16x4096 u32 keys (256KB) + A tiles + per-wave histograms (~273KB)
// -------------------------------------------------------------------------
#define A_STRIDE 136   // padded row stride (u16 elems) to dodge LDS bank conflicts
#define SMEM_BYTES (16 * NCODES * 4 + 2 * 16 * A_STRIDE * 2 + 16 * 4 + 8 * 256 * 4 + 16 * 4)

__global__ __launch_bounds__(256, 1)
void anomaly_select_kernel(const float* __restrict__ z,
                           const float* __restrict__ cbOrig,
                           const unsigned short* __restrict__ cbHi,
                           const unsigned short* __restrict__ cbLo,
                           const float* __restrict__ cbNorm,
                           const unsigned* __restrict__ cnt,
                           const unsigned* __restrict__ list,
                           float* __restrict__ outBase,
                           int HW, unsigned salt, int skip, int width) {
  extern __shared__ unsigned char smem_raw[];
  unsigned*       dist = (unsigned*)smem_raw;                     // 16*4096 keys
  unsigned short* aHi  = (unsigned short*)(dist + 16 * NCODES);   // 16*A_STRIDE
  unsigned short* aLo  = aHi + 16 * A_STRIDE;
  unsigned*       pids = (unsigned*)(aLo + 16 * A_STRIDE);        // 16
  unsigned*       hist = pids + 16;                               // 8*256
  unsigned*       bc   = hist + 8 * 256;                          // 16

  const int lane = threadIdx.x & 31;
  const int wave = threadIdx.x >> 5;

  const unsigned count = cnt[0];
  const unsigned base  = blockIdx.x * 16u;
  if (base >= count) return;   // uniform early-exit for excess blocks

  if (threadIdx.x < 16) {
    unsigned slot = base + threadIdx.x;
    if (slot >= count) slot = count - 1;  // pad with last pixel (no write later)
    pids[threadIdx.x] = list[slot];
  }
  __syncthreads();

  // stage z tile -> LDS as bf16 hi/lo (16 pixels x 128 channels, channel-planar gather)
  for (int e = threadIdx.x; e < 16 * CDIM; e += 256) {
    int pix = e >> 7, c = e & 127;
    unsigned pid = pids[pix];
    unsigned b = pid / (unsigned)HW, p = pid % (unsigned)HW;
    float v = z[((size_t)b * CDIM + c) * HW + p];
    unsigned u  = __float_as_uint(v);
    unsigned hb = u & 0xFFFF0000u;
    float    lo = v - __uint_as_float(hb);
    aHi[pix * A_STRIDE + c] = (unsigned short)(hb >> 16);
    aLo[pix * A_STRIDE + c] = (unsigned short)(__float_as_uint(lo) >> 16);
  }
  __syncthreads();

  // ---- GEMM phase: score[n] = ||cb_n||^2 - 2 * z.cb_n  (||z||^2 is rank-invariant)
  {
    const int m = lane & 15, h = lane >> 4;
    const uint4* aHiQ = (const uint4*)aHi;
    const uint4* aLoQ = (const uint4*)aLo;
    for (int t = wave; t < NCODES / 16; t += 8) {
      const int n = t * 16 + m;  // B column owned by this lane
      union { v16bf v; uint4 q[2]; } fahi, falo, fbhi, fblo;
      v8f acc = {};
      const uint4* bHiQ = (const uint4*)(cbHi + (size_t)n * CDIM);
      const uint4* bLoQ = (const uint4*)(cbLo + (size_t)n * CDIM);
#pragma unroll
      for (int kc = 0; kc < 4; ++kc) {
        // A 16x32 bf16 layout: lane half h holds K = 8h..8h+7 and 16+8h..16+8h+7
        int a1 = (m * A_STRIDE + kc * 32 + 8 * h) >> 3;
        int a2 = (m * A_STRIDE + kc * 32 + 16 + 8 * h) >> 3;
        fahi.q[0] = aHiQ[a1]; fahi.q[1] = aHiQ[a2];
        falo.q[0] = aLoQ[a1]; falo.q[1] = aLoQ[a2];
        // B 32x16 bf16 layout: lane half h holds K = 8h..8h+7 and 16+8h..16+8h+7
        int bq = (kc * 32 + 8 * h) >> 3;
        fbhi.q[0] = bHiQ[bq]; fbhi.q[1] = bHiQ[bq + 2];
        fblo.q[0] = bLoQ[bq]; fblo.q[1] = bLoQ[bq + 2];
        acc = __builtin_amdgcn_wmma_f32_16x16x32_bf16(false, fahi.v, false, fbhi.v,
                                                      (short)0, acc, false, false);
        acc = __builtin_amdgcn_wmma_f32_16x16x32_bf16(false, fahi.v, false, fblo.v,
                                                      (short)0, acc, false, false);
        acc = __builtin_amdgcn_wmma_f32_16x16x32_bf16(false, falo.v, false, fbhi.v,
                                                      (short)0, acc, false, false);
      }
      float cn = cbNorm[n];
#pragma unroll
      for (int r = 0; r < 8; ++r) {  // C/D layout: row M = r + 8*h, col N = lane&15
        float s = cn - 2.0f * acc[r];
        dist[(r + 8 * h) * NCODES + n] = f2ord(s);
      }
    }
  }
  __syncthreads();

  // ---- selection phase: wave w handles pixels w and w+8
  unsigned* whist = hist + wave * 256;
#pragma unroll 1
  for (int mi = 0; mi < 2; ++mi) {
    int m = wave + mi * 8;
    unsigned slot = base + (unsigned)m;
    if (slot >= count) continue;     // wave-uniform
    unsigned pid = pids[m];
    const unsigned* keys = dist + (size_t)m * NCODES;
    // target rank = skip + rand_col  (deterministic pixel-id hash stands in for threefry)
    int rem = skip + (int)(hash_u32(pid ^ salt) % (unsigned)width);
    unsigned prefVal = 0, prefMask = 0;
    for (int shift = 24; shift >= 0; shift -= 8) {
      for (int i = lane; i < 256; i += 32) whist[i] = 0;
      asm volatile("s_wait_dscnt 0" ::: "memory");
      for (int i = lane; i < NCODES; i += 32) {
        unsigned key = keys[i];
        if ((key & prefMask) == prefVal)
          atomicAdd(&whist[(key >> shift) & 0xFFu], 1u);
      }
      asm volatile("s_wait_dscnt 0" ::: "memory");
      if (lane == 0) {
        int run = 0; unsigned d = 0;
        for (; d < 255u; ++d) {
          int c = (int)whist[d];
          if (run + c > rem) break;
          run += c;
        }
        bc[wave * 2]     = d;
        bc[wave * 2 + 1] = (unsigned)(rem - run);
      }
      asm volatile("s_wait_dscnt 0" ::: "memory");
      unsigned d = bc[wave * 2];
      rem = (int)bc[wave * 2 + 1];
      prefVal |= d << shift;
      prefMask |= 0xFFu << shift;
    }
    // among equal keys pick ascending index (top_k stable order)
    int nstar = -1;
    for (int b0 = 0; b0 < NCODES && nstar < 0; b0 += 32) {
      bool mt = (keys[b0 + lane] == prefVal);
      unsigned mk = (unsigned)__ballot(mt);
      int c = __popc(mk);
      if (rem < c) {
        unsigned tmp = mk;
        for (int j = 0; j < rem; ++j) tmp &= tmp - 1;
        nstar = b0 + (__ffs(tmp) - 1);
      } else rem -= c;
    }
    if (nstar < 0) nstar = NCODES - 1;  // safety net
    // scatter chosen code row into masked pixel (channel-planar)
    unsigned b = pid / (unsigned)HW, p = pid % (unsigned)HW;
    const float* crow = cbOrig + (size_t)nstar * CDIM;
    float* obase = outBase + (size_t)b * CDIM * HW + p;
    for (int c = lane; c < CDIM; c += 32) obase[(size_t)c * HW] = crow[c];
  }
}

// -------------------------------------------------------------------------
// host launcher
// -------------------------------------------------------------------------
extern "C" void kernel_launch(void* const* d_in, const int* in_sizes, int n_in,
                              void* d_out, int out_size, void* d_ws, size_t ws_size,
                              hipStream_t stream) {
  (void)in_sizes; (void)n_in; (void)out_size; (void)ws_size;
  const float* q_fine    = (const float*)d_in[0];
  const float* q_coarse  = (const float*)d_in[1];
  const float* M         = (const float*)d_in[2];
  const float* cb_fine   = (const float*)d_in[3];
  const float* cb_coarse = (const float*)d_in[4];
  const float* z_fine    = (const float*)d_in[5];
  const float* z_coarse  = (const float*)d_in[6];
  float* out = (float*)d_out;

  unsigned char* ws = (unsigned char*)d_ws;
  // workspace layout
  unsigned*       cntF  = (unsigned*)(ws + 0);
  unsigned*       cntC  = cntF + 1;
  unsigned*       listF = (unsigned*)(ws + 256);
  unsigned*       listC = (unsigned*)(ws + 256 + 131072);
  unsigned short* hiF   = (unsigned short*)(ws + 164096);
  unsigned short* loF   = (unsigned short*)(ws + 1212672);
  unsigned short* hiC   = (unsigned short*)(ws + 2261248);
  unsigned short* loC   = (unsigned short*)(ws + 3309824);
  float*          normF = (float*)(ws + 4358400);
  float*          normC = (float*)(ws + 4374784);

  hipMemsetAsync(ws, 0, 256, stream);  // zero compaction counters

  const int totalQ4 = (BATCH * CDIM * 64 * 64 + BATCH * CDIM * 32 * 32) / 4;  // 1310720
  copy_q_kernel<<<totalQ4 / 256, 256, 0, stream>>>(
      (const float4*)q_fine, (const float4*)q_coarse, (float4*)out);

  pool_compact_kernel<<<(BATCH * 64 * 64 + BATCH * 32 * 32 + 255) / 256, 256, 0, stream>>>(
      M, cntF, listF, cntC, listC);

  cb_prep_kernel<<<2 * NCODES, 128, 0, stream>>>(
      cb_fine, cb_coarse, hiF, loF, normF, hiC, loC, normC);

  // skip = int(0.05*N) = 204 ; topk = N/2+1 = 2049 ; width = topk - skip = 1845
  const int skip = 204, width = 1845;
  size_t shmem = SMEM_BYTES;
  hipFuncSetAttribute(reinterpret_cast<const void*>(anomaly_select_kernel),
                      hipFuncAttributeMaxDynamicSharedMemorySize, (int)shmem);

  anomaly_select_kernel<<<(BATCH * 64 * 64) / 16, 256, shmem, stream>>>(
      z_fine, cb_fine, hiF, loF, normF, cntF, listF,
      out, 64 * 64, 0xA5A5D00Du, skip, width);

  anomaly_select_kernel<<<(BATCH * 32 * 32) / 16, 256, shmem, stream>>>(
      z_coarse, cb_coarse, hiC, loC, normC, cntC, listC,
      out + (size_t)BATCH * CDIM * 64 * 64, 32 * 32, 0x5A5AC0DEu, skip, width);
}
